// ModernBertAttention_88768384073816
// MI455X (gfx1250) — matrix operand order repaired and checked by reading, not compiled, not run on hardware
//
#include <hip/hip_runtime.h>
#include <hip/hip_bf16.h>

// ModernBERT local attention for MI455X (gfx1250, wave32, WMMA, TDM).
// Pipeline: f32->f16 casts -> TDM+LDS WMMA GEMM (QKV) -> RoPE/split ->
// windowed flash-attention with WMMA (S^T = K Q^T, O^T = V^T P) ->
// TDM+LDS WMMA GEMM (out proj).

typedef __attribute__((ext_vector_type(16))) _Float16 v16h;
typedef __attribute__((ext_vector_type(8)))  _Float16 v8h;
typedef __attribute__((ext_vector_type(8)))  float    v8f;

typedef unsigned int u32;
typedef __attribute__((ext_vector_type(4))) u32 u32x4;
typedef __attribute__((ext_vector_type(8))) int i32x8;
typedef __attribute__((ext_vector_type(4))) int i32x4;

#define NHEADS   12
#define HDIM     64
#define DMODEL   768
#define WINDOW   64
#define NEG_BIG  (-3.0e38f)

__device__ __forceinline__ v16h combine16(v8h lo, v8h hi) {
  v16h r;
#pragma unroll
  for (int i = 0; i < 8; ++i) { r[i] = lo[i]; r[i + 8] = hi[i]; }
  return r;
}

__device__ __forceinline__ v16h load16(const _Float16* p) {
  const v8h* q = (const v8h*)p;
  return combine16(q[0], q[1]);
}

__device__ __forceinline__ v8f wmma_f16(v16h a, v16h b, v8f c) {
  // v_wmma_f32_16x16x32_f16: D = A(16x32 f16) * B(32x16 f16) + C(16x16 f32)
  return __builtin_amdgcn_wmma_f32_16x16x32_f16(false, a, false, b,
                                                (short)0, c, false, false);
}

// ---- Tensor Data Mover: 2D tile (f16) global -> LDS.  D# per ISA ch.8.
// group0: count=1 | lds_addr | global_addr(57b) | type=2 ("image")
// group1: data_size=1(2B) | tensor_dim0/1 | tile_dim0/1 (tile_dim2=0) | stride0
// 6-arg builtin form (clang-23): (g0, g1, g2, g3, g4, cpol)
__device__ __forceinline__ void tdm_load_2d(u32 lds_addr, const _Float16* gptr,
                                            u32 tile_d0, u32 tile_d1,
                                            u32 tensor_d0, u32 tensor_d1,
                                            u32 stride_elems) {
  unsigned long long ga = (unsigned long long)(uintptr_t)gptr;
  u32x4 g0;
  g0[0] = 1u;                               // count=1, user descriptor
  g0[1] = lds_addr;                         // LDS byte address
  g0[2] = (u32)(ga & 0xFFFFFFFFu);          // global_addr[31:0]
  g0[3] = (u32)((ga >> 32) & 0x01FFFFFFu)   // global_addr[56:32]
        | (2u << 30);                       // type=2
  i32x8 g1;
  g1[0] = (int)(1u << 16);                  // data_size=1 -> 2 bytes
  g1[1] = (int)((tensor_d0 & 0xFFFFu) << 16);                    // dim0 lo16
  g1[2] = (int)(((tensor_d0 >> 16) & 0xFFFFu) |
                ((tensor_d1 & 0xFFFFu) << 16));                  // dim0 hi/dim1 lo
  g1[3] = (int)(((tensor_d1 >> 16) & 0xFFFFu) | (tile_d0 << 16));// dim1 hi/tile0
  g1[4] = (int)(tile_d1 & 0xFFFFu);         // tile_dim1; tile_dim2 = 0
  g1[5] = (int)stride_elems;                // tensor_dim0_stride[31:0]
  g1[6] = 0;                                // stride0[47:32], dim1_stride lo = 0
  g1[7] = 0;
  i32x4 z4 = {0, 0, 0, 0};                  // groups 2/3 unused (2-D tensor)
  i32x8 z8 = {0, 0, 0, 0, 0, 0, 0, 0};
  __builtin_amdgcn_tensor_load_to_lds(g0, g1, z4, z4, z8, 0);
}

// ---------------------------------------------------------------- f32 -> f16
__global__ void k_cvt_f16(const float* __restrict__ src,
                          _Float16* __restrict__ dst, long n) {
  long i = (long)blockIdx.x * blockDim.x + threadIdx.x;
  long stride = (long)gridDim.x * blockDim.x;
  for (; i < n; i += stride) dst[i] = (_Float16)src[i];
}

// ------------------------------------- GEMM: out = A * W^T  (TDM + LDS tiles)
// A: [M,K] f16 row-major; W: [N,K] f16 row-major.
// Block = 256 thr (8 waves, 4x2) -> 128x64 output tile; K stepped by 32.
#define MBLK 128
#define NBLK 64

__global__ __launch_bounds__(256)
void k_gemm_xwt(const _Float16* __restrict__ A, const _Float16* __restrict__ W,
                _Float16* __restrict__ outh, float* __restrict__ outf,
                int M, int N, int K) {
  __shared__ _Float16 sA[2][MBLK * 32];     // 16 KB
  __shared__ _Float16 sB[2][NBLK * 32];     //  8 KB
  int tiles_n = N / NBLK;
  int m0 = (blockIdx.x / tiles_n) * MBLK;
  int n0 = (blockIdx.x % tiles_n) * NBLK;
  int tid = threadIdx.x;
  int wave = tid >> 5, lane = tid & 31;
  int wm = wave >> 1, wn = wave & 1;        // 4 (M) x 2 (N) wave grid
  int half = lane >> 4, ln = lane & 15;

  v8f acc[2][2] = {};
  bool issuer = (tid < 32);                 // wave 0 drives the TDM
  if (issuer) {
    tdm_load_2d((u32)(uintptr_t)&sA[0][0], A + (size_t)m0 * K,
                32, MBLK, (u32)K, (u32)M, (u32)K);
    tdm_load_2d((u32)(uintptr_t)&sB[0][0], W + (size_t)n0 * K,
                32, NBLK, (u32)K, (u32)N, (u32)K);
  }
  int cur = 0;
  for (int k0 = 0; k0 < K; k0 += 32) {
    if (issuer) __builtin_amdgcn_s_wait_tensorcnt(0);
    __syncthreads();                        // buf[cur] ready for all waves
    if (issuer && (k0 + 32) < K) {          // prefetch next K-slab (dbl buffer)
      tdm_load_2d((u32)(uintptr_t)&sA[cur ^ 1][0],
                  A + (size_t)m0 * K + k0 + 32, 32, MBLK, (u32)K, (u32)M, (u32)K);
      tdm_load_2d((u32)(uintptr_t)&sB[cur ^ 1][0],
                  W + (size_t)n0 * K + k0 + 32, 32, NBLK, (u32)K, (u32)N, (u32)K);
    }
    const _Float16* As = &sA[cur][0];
    const _Float16* Bs = &sB[cur][0];
    v16h afrag[2], bfrag[2];
#pragma unroll
    for (int mi = 0; mi < 2; ++mi) {        // A 16x32: 32B contiguous / lane
      const _Float16* ar = As + (size_t)(wm * 32 + mi * 16 + ln) * 32;
      afrag[mi] = load16(ar + half * 16);
    }
#pragma unroll
    for (int ni = 0; ni < 2; ++ni) {        // B 32x16: 32B contiguous / lane
      const _Float16* br = Bs + (size_t)(wn * 32 + ni * 16 + ln) * 32;
      bfrag[ni] = load16(br + half * 16);
    }
#pragma unroll
    for (int mi = 0; mi < 2; ++mi)
#pragma unroll
      for (int ni = 0; ni < 2; ++ni)
        acc[mi][ni] = wmma_f16(afrag[mi], bfrag[ni], acc[mi][ni]);
    __syncthreads();                        // done reading buf[cur]
    cur ^= 1;
  }
#pragma unroll
  for (int mi = 0; mi < 2; ++mi)
#pragma unroll
    for (int ni = 0; ni < 2; ++ni)
#pragma unroll
      for (int r = 0; r < 8; ++r) {
        size_t idx = (size_t)(m0 + wm * 32 + mi * 16 + r + half * 8) * N
                   + (n0 + wn * 32 + ni * 16 + ln);
        if (outh) outh[idx] = (_Float16)acc[mi][ni][r];
        else      outf[idx] = acc[mi][ni][r];
      }
}

// ---------------------------------------- RoPE + split + V transpose
// qkv: [B,T,3*768] f16.  Writes Q,K: [B,H,T,64] (Q prescaled by 1/8),
// Vt: [B,H,64,T].  RoPE pairs (i, i+32) -> interleaved (2i, 2i+1).
__global__ void k_rope_split(const _Float16* __restrict__ qkv,
                             _Float16* __restrict__ Q,
                             _Float16* __restrict__ Kk,
                             _Float16* __restrict__ Vt,
                             int B, int T) {
  int idx = blockIdx.x * blockDim.x + threadIdx.x;
  int total = B * NHEADS * T * 32;
  if (idx >= total) return;
  int i  = idx & 31;
  int t  = (idx >> 5) % T;
  int bh = idx / (32 * T);
  int h  = bh % NHEADS;
  int b  = bh / NHEADS;

  size_t in = ((size_t)b * T + t) * (3 * DMODEL);
  float qr = (float)qkv[in + h * HDIM + i];
  float qi = (float)qkv[in + h * HDIM + i + 32];
  float kr = (float)qkv[in + DMODEL + h * HDIM + i];
  float ki = (float)qkv[in + DMODEL + h * HDIM + i + 32];
  float va = (float)qkv[in + 2 * DMODEL + h * HDIM + i];
  float vb = (float)qkv[in + 2 * DMODEL + h * HDIM + i + 32];

  float ang = (float)t * __powf(10000.0f, -(float)i / 32.0f);
  float s, c;
  __sincosf(ang, &s, &c);

  size_t ob = ((size_t)bh * T + t) * HDIM;
  Q[ob + 2 * i]     = (_Float16)(0.125f * (qr * c - qi * s)); // fold 1/sqrt(64)
  Q[ob + 2 * i + 1] = (_Float16)(0.125f * (qr * s + qi * c));
  Kk[ob + 2 * i]     = (_Float16)(kr * c - ki * s);
  Kk[ob + 2 * i + 1] = (_Float16)(kr * s + ki * c);

  size_t vbase = (size_t)bh * HDIM;
  Vt[(vbase + i) * T + t]      = (_Float16)va;
  Vt[(vbase + i + 32) * T + t] = (_Float16)vb;
}

// ------------------------------------- windowed flash attention (1 wave / 16q)
// Q,K: [B,H,T,64], Vt: [B,H,64,T].  attn out: [B,T,768] f16.
__global__ __launch_bounds__(256)
void k_attn(const _Float16* __restrict__ Q,
            const _Float16* __restrict__ Kt,
            const _Float16* __restrict__ Vt,
            const int* __restrict__ pad,
            _Float16* __restrict__ attn,
            int B, int T) {
  int wid  = (blockIdx.x * blockDim.x + threadIdx.x) >> 5;
  int lane = threadIdx.x & 31;
  int QT = T >> 4;
  int total = B * NHEADS * QT;
  if (wid >= total) return;                // wave-uniform
  int qt = wid % QT;
  int bh = wid / QT;
  int b  = bh / NHEADS;
  int half = lane >> 4, ln = lane & 15;

  const _Float16* Qh = Q  + (size_t)bh * T * HDIM;
  const _Float16* Kh = Kt + (size_t)bh * T * HDIM;
  const _Float16* Vh = Vt + (size_t)bh * HDIM * T;

  int q0  = qt << 4;
  int myq = q0 + ln;                       // this lane's query column
  bool padq = pad[(size_t)b * T + myq] != 0;

  // Q as B-fragment of S^T = K * Q^T  (two K=32 chunks of head dim)
  v16h bq0 = load16(Qh + (size_t)myq * HDIM + half * 16);
  v16h bq1 = load16(Qh + (size_t)myq * HDIM + 32 + half * 16);

  v8f o0 = {}, o1 = {}, o2 = {}, o3 = {};   // O^T accumulators (4 d-tiles)
  float m_run = NEG_BIG, l_run = 0.0f;

  for (int j = 0; j < 9; ++j) {
    int kt0 = q0 + (j - 4) * 16;           // key tile start
    if (kt0 < 0 || kt0 >= T) continue;     // wave-uniform skip

    // ---- S^T tile: D[m=key, n=query]
    const _Float16* krow = Kh + (size_t)(kt0 + ln) * HDIM;
    v16h aK0 = load16(krow + half * 16);
    v16h aK1 = load16(krow + 32 + half * 16);
    v8f s = {};
    s = wmma_f16(aK0, bq0, s);
    s = wmma_f16(aK1, bq1, s);

    // ---- mask + online softmax (row = query = lane%16)
    float sv[8];
    float mt = NEG_BIG;
#pragma unroll
    for (int r = 0; r < 8; ++r) {
      int key = kt0 + r + half * 8;
      int d = myq - key; d = d < 0 ? -d : d;
      bool ok = (d <= WINDOW) && padq && (pad[(size_t)b * T + key] != 0);
      sv[r] = ok ? s[r] : NEG_BIG;
      mt = fmaxf(mt, sv[r]);
    }
    mt = fmaxf(mt, __shfl_xor(mt, 16, 32));
    float m_new = fmaxf(m_run, mt);
    float alpha = __expf(m_run - m_new);
    m_run = m_new;
    l_run *= alpha;
#pragma unroll
    for (int r = 0; r < 8; ++r) {
      o0[r] *= alpha; o1[r] *= alpha; o2[r] *= alpha; o3[r] *= alpha;
    }
    float pr[8], ls = 0.0f;
#pragma unroll
    for (int r = 0; r < 8; ++r) { pr[r] = __expf(sv[r] - m_new); ls += pr[r]; }
    ls += __shfl_xor(ls, 16, 32);
    l_run += ls;

    // ---- build P as B-fragment (32x16, rows 16..31 zero-padded)
    float tmv[8];
#pragma unroll
    for (int r = 0; r < 8; ++r) tmv[r] = __shfl_xor(pr[r], 16, 32);
    v16h bp;
#pragma unroll
    for (int hh = 0; hh < 8; ++hh) {
      bp[hh]     = (half == 0) ? (_Float16)pr[hh]  : (_Float16)0.0f;
      bp[hh + 8] = (half == 0) ? (_Float16)tmv[hh] : (_Float16)0.0f;
    }

    // ---- O^T += V^T * P   (A from Vt, contiguous; clamp hi segment at edge)
    int koff2 = (kt0 + 32 <= T) ? (kt0 + 16) : kt0;  // safe addr; B rows are 0
    auto loadAV = [&](int dt) -> v16h {
      const _Float16* vrow = Vh + (size_t)(dt * 16 + ln) * T;
      return load16(vrow + (half ? koff2 : kt0));
    };
    o0 = wmma_f16(loadAV(0), bp, o0);
    o1 = wmma_f16(loadAV(1), bp, o1);
    o2 = wmma_f16(loadAV(2), bp, o2);
    o3 = wmma_f16(loadAV(3), bp, o3);
  }

  float inv = 1.0f / l_run;
  int hHead = bh % NHEADS;
  size_t obase = ((size_t)b * T + myq) * DMODEL + hHead * HDIM;
#pragma unroll
  for (int r = 0; r < 8; ++r) {
    int dr = r + half * 8;
    attn[obase + dr]      = (_Float16)(o0[r] * inv);
    attn[obase + 16 + dr] = (_Float16)(o1[r] * inv);
    attn[obase + 32 + dr] = (_Float16)(o2[r] * inv);
    attn[obase + 48 + dr] = (_Float16)(o3[r] * inv);
  }
}

// --------------------------------------------------------------- launcher
extern "C" void kernel_launch(void* const* d_in, const int* in_sizes, int n_in,
                              void* d_out, int out_size, void* d_ws, size_t ws_size,
                              hipStream_t stream) {
  const float* hs   = (const float*)d_in[0];   // [B,T,768]
  const int*   mask = (const int*)  d_in[1];   // [B,T]
  const float* wqkv = (const float*)d_in[2];   // [2304,768]
  const float* wo   = (const float*)d_in[3];   // [768,768]
  float* out = (float*)d_out;

  const int B = 4;
  const int T = in_sizes[1] / B;               // 2048
  const int M = B * T;                         // 8192 rows

  // workspace layout (f16 staging buffers)
  char* ws = (char*)d_ws;
  size_t off = 0;
  auto alloc = [&](size_t bytes) -> void* {
    void* p = ws + off;
    off = (off + bytes + 255) & ~(size_t)255;
    return p;
  };
  _Float16* hs_h  = (_Float16*)alloc((size_t)M * DMODEL * 2);
  _Float16* wq_h  = (_Float16*)alloc((size_t)3 * DMODEL * DMODEL * 2);
  _Float16* wo_h  = (_Float16*)alloc((size_t)DMODEL * DMODEL * 2);
  _Float16* qkv_h = (_Float16*)alloc((size_t)M * 3 * DMODEL * 2);
  _Float16* q_h   = (_Float16*)alloc((size_t)M * DMODEL * 2);
  _Float16* k_h   = (_Float16*)alloc((size_t)M * DMODEL * 2);
  _Float16* vt_h  = (_Float16*)alloc((size_t)M * DMODEL * 2);
  _Float16* at_h  = hs_h;                      // attn output reuses hs_h (dead)

  const int TB = 256;
  // 1) casts
  {
    long n = (long)M * DMODEL;
    k_cvt_f16<<<(int)((n + TB - 1) / TB), TB, 0, stream>>>(hs, hs_h, n);
    n = (long)3 * DMODEL * DMODEL;
    k_cvt_f16<<<(int)((n + TB - 1) / TB), TB, 0, stream>>>(wqkv, wq_h, n);
    n = (long)DMODEL * DMODEL;
    k_cvt_f16<<<(int)((n + TB - 1) / TB), TB, 0, stream>>>(wo, wo_h, n);
  }
  // 2) QKV = hs * Wqkv^T  (f16 out); 128x64 block tiles
  {
    int blocks = (M / MBLK) * ((3 * DMODEL) / NBLK);
    k_gemm_xwt<<<blocks, TB, 0, stream>>>(hs_h, wq_h, qkv_h, nullptr,
                                          M, 3 * DMODEL, DMODEL);
  }
  // 3) RoPE + split + V transpose
  {
    long n = (long)B * NHEADS * T * 32;
    k_rope_split<<<(int)((n + TB - 1) / TB), TB, 0, stream>>>(
        qkv_h, q_h, k_h, vt_h, B, T);
  }
  // 4) sliding-window attention
  {
    long waves = (long)B * NHEADS * (T / 16);
    long thr = waves * 32;
    k_attn<<<(int)((thr + TB - 1) / TB), TB, 0, stream>>>(
        q_h, k_h, vt_h, mask, at_h, B, T);
  }
  // 5) out = attn * Wo^T  (f32 out)
  {
    int blocks = (M / MBLK) * (DMODEL / NBLK);
    k_gemm_xwt<<<blocks, TB, 0, stream>>>(at_h, wo_h, nullptr, out,
                                          M, DMODEL, DMODEL);
  }
  (void)n_in; (void)out_size; (void)ws_size;
}
